// MSAAttention_44375602102594
// MI455X (gfx1250) — matrix operand order repaired and checked by reading, not compiled, not run on hardware
//
#include <hip/hip_runtime.h>
#include <cstdint>

// MSA row attention with gating for MI455X (gfx1250), wave32 + WMMA bf16.
// All GEMMs via v_wmma_f32_16x16x32_bf16. B operands are stored transposed
// ([N][K]) so every lane's B-fragment is one contiguous, 32B-aligned v16bf load.

typedef __bf16 bf16_t;
typedef __attribute__((ext_vector_type(16))) __bf16 v16bf;
typedef __attribute__((ext_vector_type(8)))  float  v8f;

#define S_  128
#define R_  384
#define C_  256
#define H_  8
#define D_  32
#define HD_ 256

// ---------------------------------------------------------------------------
// WMMA fragment loaders (contiguous k order chosen by this file, same for A and B, wave32)
// A 16x32 bf16: lane<16 -> row=lane,   K=0..15
//               lane>=16 -> row=lane-16, K=16..31
// B 32x16 bf16: lane<16 -> col=lane, K=0..15; lane>=16 -> col=lane-16, K=16..31
// C/D 16x16 f32: lane<16 -> col=lane rows 0..7; lane>=16 -> col=lane-16 rows 8..15
// ---------------------------------------------------------------------------
static __device__ __forceinline__ v16bf load_a_bf_lds(const bf16_t* base, int ld) {
  const int lane = threadIdx.x & 31;
  const bf16_t* rp = base + (lane & 15) * ld;
  const int kb = (lane & 16) ? 16 : 0;
  v16bf a;
#pragma unroll
  for (int v = 0; v < 8; ++v) {
    const int k = kb + v * 2;
    a[2 * v]     = rp[k];
    a[2 * v + 1] = rp[k + 1];
  }
  return a;
}

// B fragment from a transposed ([N][K] row-major) matrix: per lane one
// contiguous 16-element (32B) load -> 2x global_load_b128.
static __device__ __forceinline__ v16bf load_bT(const bf16_t* baseT, long ldT) {
  const int lane = threadIdx.x & 31;
  const bf16_t* p = baseT + (long)(lane & 15) * ldT + ((lane & 16) ? 16 : 0);
  return *(const v16bf*)p;
}

static __device__ __forceinline__ float wave_sum32(float v) {
#pragma unroll
  for (int off = 16; off > 0; off >>= 1) v += __shfl_xor(v, off, 32);
  return v;
}
static __device__ __forceinline__ float wave_max32(float v) {
#pragma unroll
  for (int off = 16; off > 0; off >>= 1) v = fmaxf(v, __shfl_xor(v, off, 32));
  return v;
}

// ---------------------------------------------------------------------------
// Kernel 0: weights f32 -> bf16, TRANSPOSED into one contiguous slab:
// wT_all = [wqT | wkT | wvT | wgT | woT], each [256][256] as [N][K].
// ---------------------------------------------------------------------------
__global__ void convert_weights(const float* __restrict__ wq, const float* __restrict__ wk,
                                const float* __restrict__ wv, const float* __restrict__ wg,
                                const float* __restrict__ wo, bf16_t* __restrict__ wT_all) {
  const int i = blockIdx.x * 256 + threadIdx.x;   // i = k*256 + n (coalesced reads)
  const int W = C_ * HD_;
  if (i < W) {
    const int k = i >> 8, n = i & 255;
    const int t = n * 256 + k;
    wT_all[t]         = (bf16_t)wq[i];
    wT_all[W + t]     = (bf16_t)wk[i];
    wT_all[2 * W + t] = (bf16_t)wv[i];
    wT_all[3 * W + t] = (bf16_t)wg[i];
    wT_all[4 * W + t] = (bf16_t)wo[i];
  }
}

// ---------------------------------------------------------------------------
// Kernel 1: LayerNorm + Q/K/V/G projections, 64 rows per block.
// grid (S_, R_/64), block 256 (8 waves).
// q,k stored head-major [ (s*H+h)*R + r ]*D + d (q pre-scaled by D^-0.5),
// v stored TRANSPOSED per head: vT[ ((s*H+h)*D + d)*R + r ],
// g = sigmoid(xWg + bg) head-major like q/k.
// ---------------------------------------------------------------------------
#define XLD 272
__global__ void ln_qkvg(const float* __restrict__ m,
                        const float* __restrict__ ln_w, const float* __restrict__ ln_b,
                        const float* __restrict__ bg,
                        const bf16_t* __restrict__ wT_all,
                        bf16_t* __restrict__ qh, bf16_t* __restrict__ kh,
                        bf16_t* __restrict__ vT, bf16_t* __restrict__ gh) {
  __shared__ bf16_t xb[64][XLD];
  const int s  = blockIdx.x;
  const int r0 = blockIdx.y * 64;
  const int w    = threadIdx.x >> 5;
  const int lane = threadIdx.x & 31;

  // ---- LayerNorm: wave w handles rows 8w .. 8w+7 ----
  for (int rr = 8 * w; rr < 8 * w + 8; ++rr) {
    const float* mp = m + ((long)s * R_ + (r0 + rr)) * C_;
    float vals[8];
    float sum = 0.f, sq = 0.f;
#pragma unroll
    for (int j = 0; j < 8; ++j) {
      const float x = mp[lane + 32 * j];
      vals[j] = x; sum += x; sq += x * x;
    }
    sum = wave_sum32(sum);
    sq  = wave_sum32(sq);
    const float mu   = sum * (1.0f / C_);
    const float var  = sq * (1.0f / C_) - mu * mu;
    const float rstd = rsqrtf(var + 1e-5f);
#pragma unroll
    for (int j = 0; j < 8; ++j) {
      const int c = lane + 32 * j;
      xb[rr][c] = (bf16_t)((vals[j] - mu) * rstd * ln_w[c] + ln_b[c]);
    }
  }
  __syncthreads();

  const int col_l = lane & 15;
  const int rbase = (lane & 16) ? 8 : 0;
  const size_t W = (size_t)C_ * HD_;

  // ---- 64 col-tiles (4 matrices x 16), 8 per wave; B reused over 4 row-tiles
#pragma unroll
  for (int i = 0; i < 8; ++i) {
    const int colt = w + i * 8;          // 0..63
    const int mi   = colt >> 4;          // which matrix
    const int n0   = (colt & 15) * 16;   // output col block
    const bf16_t* WT = wT_all + (size_t)mi * W;
    v8f acc[4] = {};
#pragma unroll
    for (int k0 = 0; k0 < C_; k0 += 32) {
      const v16bf b = load_bT(WT + (long)n0 * C_ + k0, C_);
#pragma unroll
      for (int rt = 0; rt < 4; ++rt) {
        v16bf a = load_a_bf_lds(&xb[rt * 16][0] + k0, XLD);
        acc[rt] = __builtin_amdgcn_wmma_f32_16x16x32_bf16(false, a, false, b,
                                                          (short)0, acc[rt], false, false);
      }
    }
    const int col = n0 + col_l;
    const int hh = col >> 5, dd = col & 31;
    const long hb  = ((long)(s * H_ + hh) * R_) * D_;
    const long vtb = ((long)(s * H_ + hh) * D_ + dd) * R_;
#pragma unroll
    for (int rt = 0; rt < 4; ++rt) {
      const int rbe = r0 + rt * 16 + rbase;
      if (mi == 0) {
#pragma unroll
        for (int v = 0; v < 8; ++v)
          qh[hb + (long)(rbe + v) * D_ + dd] = (bf16_t)(acc[rt][v] * 0.17677669529663687f);
      } else if (mi == 1) {
#pragma unroll
        for (int v = 0; v < 8; ++v)
          kh[hb + (long)(rbe + v) * D_ + dd] = (bf16_t)acc[rt][v];
      } else if (mi == 2) {
#pragma unroll
        for (int v = 0; v < 8; ++v)
          vT[vtb + (rbe + v)] = (bf16_t)acc[rt][v];
      } else {
        const float bgl = bg[col];
#pragma unroll
        for (int v = 0; v < 8; ++v)
          gh[hb + (long)(rbe + v) * D_ + dd] =
              (bf16_t)(1.0f / (1.0f + __expf(-(acc[rt][v] + bgl))));
      }
    }
  }
}

// ---------------------------------------------------------------------------
// Kernel 2: attention per (s, h, 64 q-rows). grid (S_*H_, R_/64), block 256.
// scores = q @ k^T + mask_bias (bf16 in LDS) -> softmax (bf16 probs in LDS)
// -> o = P @ v, one 16x16 tile per wave, full K=384 accumulation chain.
// q tile staged via global_load_async_to_lds_b128 (ASYNCcnt path).
// ---------------------------------------------------------------------------
#define PLD 400
__global__ void attn(const bf16_t* __restrict__ qh, const bf16_t* __restrict__ kh,
                     const bf16_t* __restrict__ vT, const float* __restrict__ mask,
                     bf16_t* __restrict__ oh) {
  __shared__ bf16_t qs[64][D_];          // 4 KB, contiguous for async copy
  __shared__ bf16_t scb[64][PLD];        // 50 KB: scores, then exp() probs
  __shared__ float  rs[64];

  const int s  = blockIdx.x >> 3;
  const int h  = blockIdx.x & 7;
  const int r0 = blockIdx.y * 64;
  const int tid  = threadIdx.x;
  const int w    = tid >> 5;
  const int lane = tid & 31;
  const long hb  = ((long)(s * H_ + h) * R_) * D_;

  // ---- async-copy q tile (contiguous 4096 B): every thread one b128 ----
  {
    const unsigned ldsoff = (unsigned)(uintptr_t)(&qs[0][0]) + (unsigned)tid * 16u;
    const bf16_t* gsrc = qh + hb + (long)r0 * D_ + tid * 8;
    asm volatile("global_load_async_to_lds_b128 %0, %1, off"
                 :: "v"(ldsoff), "v"(gsrc) : "memory");
  }
  asm volatile("s_wait_asynccnt 0" ::: "memory");
  __syncthreads();

  const int col_l = lane & 15;
  const int rbase = (lane & 16) ? 8 : 0;

  // ---- q fragments for 4 row-tiles, kept in registers ----
  v16bf aq[4];
#pragma unroll
  for (int rt = 0; rt < 4; ++rt) aq[rt] = load_a_bf_lds(&qs[rt * 16][0], D_);

  // ---- scores: 24 key tiles, wave w does ct = w, w+8, w+16; 4 row-tiles each
#pragma unroll
  for (int ct = w; ct < 24; ct += 8) {
    const int n0k = ct * 16;
    if (ct + 8 < 24)
      __builtin_prefetch(kh + hb + (long)(n0k + 128) * D_, 0, 0);  // global_prefetch_b8
    const v16bf b = load_bT(kh + hb + (long)n0k * D_, D_);         // k^T: contiguous in d
    const int key  = n0k + col_l;
    const float bias = 1e9f * (mask[(long)s * R_ + key] - 1.0f);
#pragma unroll
    for (int rt = 0; rt < 4; ++rt) {
      v8f acc = {};
      acc = __builtin_amdgcn_wmma_f32_16x16x32_bf16(false, aq[rt], false, b,
                                                    (short)0, acc, false, false);
#pragma unroll
      for (int v = 0; v < 8; ++v)
        scb[rt * 16 + rbase + v][key] = (bf16_t)(acc[v] + bias);
    }
  }
  __syncthreads();

  // ---- softmax (exp stored bf16 + f32 row sums): wave w -> rows 8w..8w+7 ----
  for (int rr = 8 * w; rr < 8 * w + 8; ++rr) {
    float e[12];
    float mx = -1e30f;
#pragma unroll
    for (int j = 0; j < 12; ++j) {
      e[j] = (float)scb[rr][lane + 32 * j];
      mx = fmaxf(mx, e[j]);
    }
    mx = wave_max32(mx);
    float sum = 0.f;
#pragma unroll
    for (int j = 0; j < 12; ++j) {
      e[j] = __expf(e[j] - mx);
      sum += e[j];
      scb[rr][lane + 32 * j] = (bf16_t)e[j];
    }
    sum = wave_sum32(sum);
    if (lane == 0) rs[rr] = sum;
  }
  __syncthreads();

  // ---- PV: one 16x16 tile per wave (rt = w>>1, ct = w&1), 12 WMMAs K=384 ----
  {
    const int rt = w >> 1;
    const int ct = w & 1;
    const long vtb = ((long)(s * H_ + h) * D_ + ct * 16) * R_;
    v8f acc = {};
#pragma unroll
    for (int c12 = 0; c12 < 12; ++c12) {
      const int kk0 = c12 * 32;
      v16bf a = load_a_bf_lds(&scb[rt * 16][0] + kk0, PLD);
      v16bf b = load_bT(vT + vtb + kk0, R_);   // v^T: contiguous in r
      acc = __builtin_amdgcn_wmma_f32_16x16x32_bf16(false, a, false, b,
                                                    (short)0, acc, false, false);
    }
    // normalize + store o (bf16, head-major)
    const int col = ct * 16 + col_l;
#pragma unroll
    for (int v = 0; v < 8; ++v) {
      const int rr = rt * 16 + rbase + v;
      oh[hb + (long)(r0 + rr) * D_ + col] = (bf16_t)(acc[v] / rs[rr]);
    }
  }
}

// ---------------------------------------------------------------------------
// Kernel 3: out = (o * g) @ wo + bo, 64 rows per block. grid (S_, R_/64).
// ---------------------------------------------------------------------------
__global__ void gate_out(const bf16_t* __restrict__ oh, const bf16_t* __restrict__ gh,
                         const bf16_t* __restrict__ woT, const float* __restrict__ bo,
                         float* __restrict__ out) {
  __shared__ bf16_t ab[64][XLD];
  const int s  = blockIdx.x;
  const int r0 = blockIdx.y * 64;
  const int tid  = threadIdx.x;
  const int w    = tid >> 5;
  const int lane = tid & 31;

  // build gated activation tile (64 x 256) in LDS
  for (int idx = tid; idx < 64 * HD_; idx += 256) {
    const int rr = idx >> 8, cc = idx & 255;
    const int hh = cc >> 5, dd = cc & 31;
    const long p = ((long)(s * H_ + hh) * R_ + (r0 + rr)) * D_ + dd;
    ab[rr][cc] = (bf16_t)((float)oh[p] * (float)gh[p]);
  }
  __syncthreads();

  const int col_l = lane & 15;
  const int rbase = (lane & 16) ? 8 : 0;

#pragma unroll
  for (int i = 0; i < 2; ++i) {
    const int colt = w + i * 8;          // 16 output col tiles
    const int n0 = colt * 16;
    v8f acc[4] = {};
#pragma unroll
    for (int k0 = 0; k0 < HD_; k0 += 32) {
      const v16bf b = load_bT(woT + (long)n0 * HD_ + k0, HD_);
#pragma unroll
      for (int rt = 0; rt < 4; ++rt) {
        v16bf a = load_a_bf_lds(&ab[rt * 16][0] + k0, XLD);
        acc[rt] = __builtin_amdgcn_wmma_f32_16x16x32_bf16(false, a, false, b,
                                                          (short)0, acc[rt], false, false);
      }
    }
    const int col = n0 + col_l;
    const float bias = bo[col];
#pragma unroll
    for (int rt = 0; rt < 4; ++rt) {
      const int rbe = r0 + rt * 16 + rbase;
#pragma unroll
      for (int v = 0; v < 8; ++v)
        out[((long)s * R_ + (rbe + v)) * C_ + col] = acc[rt][v] + bias;
    }
  }
}

// ---------------------------------------------------------------------------
extern "C" void kernel_launch(void* const* d_in, const int* in_sizes, int n_in,
                              void* d_out, int out_size, void* d_ws, size_t ws_size,
                              hipStream_t stream) {
  (void)in_sizes; (void)n_in; (void)out_size; (void)ws_size;
  const float* m    = (const float*)d_in[0];
  const float* mask = (const float*)d_in[1];
  const float* ln_w = (const float*)d_in[2];
  const float* ln_b = (const float*)d_in[3];
  const float* wq   = (const float*)d_in[4];
  const float* wk   = (const float*)d_in[5];
  const float* wv   = (const float*)d_in[6];
  const float* wg   = (const float*)d_in[7];
  const float* bg   = (const float*)d_in[8];
  const float* wo   = (const float*)d_in[9];
  const float* bo   = (const float*)d_in[10];
  float* out = (float*)d_out;

  bf16_t* ws = (bf16_t*)d_ws;
  const size_t W = (size_t)C_ * HD_;                 // 65536
  const size_t P = (size_t)S_ * H_ * R_ * D_;        // 12,582,912
  bf16_t* wT_all = ws;                               // 5*W (wq,wk,wv,wg,wo transposed)
  bf16_t* woT = wT_all + 4 * W;
  bf16_t* qh  = wT_all + 5 * W;
  bf16_t* kh  = qh + P;
  bf16_t* vT  = kh + P;
  bf16_t* gh  = vT + P;
  bf16_t* oh  = gh + P;

  convert_weights<<<(int)((W + 255) / 256), 256, 0, stream>>>(wq, wk, wv, wg, wo, wT_all);
  ln_qkvg<<<dim3(S_, R_ / 64), 256, 0, stream>>>(m, ln_w, ln_b, bg, wT_all,
                                                 qh, kh, vT, gh);
  attn<<<dim3(S_ * H_, R_ / 64), 256, 0, stream>>>(qh, kh, vT, mask, oh);
  gate_out<<<dim3(S_, R_ / 64), 256, 0, stream>>>(oh, gh, woT, bo, out);
}
